// ConditionalUnet1D_20435454394458
// MI455X (gfx1250) — matrix operand order repaired and checked by reading, not compiled, not run on hardware
//
#include <hip/hip_runtime.h>
#include <hip/hip_bf16.h>
#include <math.h>

#define NLAY 6
#define N_   4
#define C_   256
#define CC_  256
#define L_   4096
#define T_   512
#define KS_  5
#define NL_  (N_ * L_)

typedef __attribute__((ext_vector_type(16))) __bf16          v16bf;
typedef __attribute__((ext_vector_type(8)))  float           v8f;
typedef __attribute__((ext_vector_type(8)))  unsigned short  v8us;
typedef unsigned short u16;

union ABReg { v16bf v; v8us u[2]; };

__device__ __forceinline__ u16 f2bf(float f) {
  unsigned u = __float_as_uint(f);
  u += 0x7FFFu + ((u >> 16) & 1u);   // round-to-nearest-even
  return (u16)(u >> 16);
}

// ---------------------------------------------------------------------------
// K0: x += (swish(time_emb) @ time_w^T + time_b), transposed write to X[N,L,C]
// ---------------------------------------------------------------------------
__global__ __launch_bounds__(256)
void time_kernel(const float* __restrict__ x, const float* __restrict__ te,
                 const float* __restrict__ tw, const float* __restrict__ tb,
                 float* __restrict__ X) {
  __shared__ float red[256];
  const int tid = threadIdx.x;
  const int n = blockIdx.x >> 8;
  const int c = blockIdx.x & 255;
  float s = 0.f;
  for (int t = tid; t < T_; t += 256) {
    float e = te[n * T_ + t];
    float sw = e / (1.f + expf(-e));
    s += sw * tw[c * T_ + t];
  }
  red[tid] = s;
  __syncthreads();
  for (int st = 128; st > 0; st >>= 1) {
    if (tid < st) red[tid] += red[tid + st];
    __syncthreads();
  }
  const float bias = red[0] + tb[c];
  const long src = ((long)n * C_ + c) * L_;
  const long dst = (long)n * L_ * C_ + c;
  for (int l = tid; l < L_; l += 256)
    X[dst + (long)l * C_] = x[src + l] + bias;
}

// elementwise f32 -> bf16
__global__ __launch_bounds__(256)
void cvt_kernel(const float* __restrict__ src, u16* __restrict__ dst, int count) {
  int i = blockIdx.x * 256 + threadIdx.x;
  if (i < count) dst[i] = f2bf(src[i]);
}

// x_cross [N,CC,L] f32 -> xct [N,L,CC] bf16
__global__ __launch_bounds__(256)
void xct_kernel(const float* __restrict__ xc, u16* __restrict__ xct) {
  int i = blockIdx.x * 256 + threadIdx.x;
  if (i >= N_ * CC_ * L_) return;
  int n = i / (CC_ * L_);
  int c = (i / L_) % CC_;
  int l = i % L_;
  xct[((long)n * L_ + l) * CC_ + c] = f2bf(xc[i]);
}

// conv_w [nL,C,C,KS] f32 -> tap-major bf16 [nL][KS][C][C]
__global__ __launch_bounds__(256)
void convw_kernel(const float* __restrict__ cw, u16* __restrict__ W) {
  int i = blockIdx.x * 256 + threadIdx.x;
  if (i >= NLAY * KS_ * C_ * C_) return;
  int layer = i / (KS_ * C_ * C_);
  int rem   = i % (KS_ * C_ * C_);
  int j  = rem / (C_ * C_);
  int r2 = rem % (C_ * C_);
  int o  = r2 / C_;
  int c  = r2 % C_;
  W[i] = f2bf(cw[(((long)(layer * C_ + o) * C_ + c) * KS_) + j]);
}

// X [N,L,C] f32 -> d_out [N,C,L] f32
__global__ __launch_bounds__(256)
void outT_kernel(const float* __restrict__ X, float* __restrict__ out) {
  int i = blockIdx.x * 256 + threadIdx.x;
  if (i >= N_ * C_ * L_) return;
  int n = i / (C_ * L_);
  int c = (i / L_) & 255;
  int l = i & (L_ - 1);
  out[i] = X[((long)n * L_ + l) * C_ + c];
}

__device__ __forceinline__ v8f wmma_bf16(const ABReg& a, const ABReg& b, v8f c) {
  return __builtin_amdgcn_wmma_f32_16x16x32_bf16(false, a.v, false, b.v,
                                                 (short)0, c, false, false);
}

// ---------------------------------------------------------------------------
// Pointwise WMMA GEMM: D[256 x NL] = A[256,K0+K1] x [B0;B1]
// wave tile 64(M) x 32(N), block tile 256 x 64, grid NL/64.
// Fully-unrolled, statically software-pipelined K loop (no register rotation).
// MODE 0: outF = D + bias (f32 [N,L,C])
// MODE 1: outB = bf16(relu(D + bias))
// MODE 2: resid += D + bias
// ---------------------------------------------------------------------------
template <int K0, int K1, int MODE>
__global__ __launch_bounds__(256)
void gemm_pw(const u16* __restrict__ A, const u16* __restrict__ B0,
             const u16* __restrict__ B1, const float* __restrict__ bias,
             float* __restrict__ outF, u16* __restrict__ outB,
             float* __restrict__ resid) {
  const int lane  = threadIdx.x & 31;
  const int wave  = threadIdx.x >> 5;
  const int half  = lane >> 4;
  const int mBase = (wave & 3) * 64;
  const int col0  = blockIdx.x * 64 + (wave >> 2) * 32 + (lane & 15);
  const int col1  = col0 + 16;
  constexpr int Ktot = K0 + K1;
  constexpr int S0 = K0 >> 5;
  constexpr int S  = S0 + (K1 >> 5);
  const int mRow  = mBase + (lane & 15);

  const u16* Arow[4];
  for (int t = 0; t < 4; ++t) Arow[t] = A + (long)(mRow + t * 16) * Ktot;
  const u16* b0p = B0 + (long)col0 * K0;
  const u16* b1p = B0 + (long)col1 * K0;
  const u16* c0p = (K1 > 0) ? (B1 + (long)col0 * K1) : nullptr;
  const u16* c1p = (K1 > 0) ? (B1 + (long)col1 * K1) : nullptr;

  v8f acc[4][2];
  for (int t = 0; t < 4; ++t)
    for (int nn = 0; nn < 2; ++nn)
      for (int r = 0; r < 8; ++r) acc[t][nn][r] = 0.f;

  auto loadStep = [&](int s, ABReg a[4], ABReg b[2]) {
    const u16 *p0, *p1;
    if (s < S0) { p0 = b0p + (s << 5); p1 = b1p + (s << 5); }
    else { const int off = (s - S0) << 5; p0 = c0p + off; p1 = c1p + off; }
    b[0].u[0] = *(const v8us*)(p0 + 16 * half);
    b[0].u[1] = *(const v8us*)(p0 + 16 * half + 8);
    b[1].u[0] = *(const v8us*)(p1 + 16 * half);
    b[1].u[1] = *(const v8us*)(p1 + 16 * half + 8);
    const int acol = s << 5;
    for (int t = 0; t < 4; ++t) {
      const u16* ar = Arow[t] + acol;
      a[t].u[0] = *(const v8us*)(ar + 16 * half);
      a[t].u[1] = *(const v8us*)(ar + 16 * half + 8);
    }
  };
  auto domma = [&](ABReg a[4], ABReg b[2]) {
    for (int t = 0; t < 4; ++t) {
      acc[t][0] = wmma_bf16(a[t], b[0], acc[t][0]);
      acc[t][1] = wmma_bf16(a[t], b[1], acc[t][1]);
    }
  };

  // statically pipelined: loads of step s+1 precede WMMAs of step s; with the
  // loop fully unrolled the (s&1) buffers become distinct SSA values (no movs).
  ABReg aBuf[2][4], bBuf[2][2];
  loadStep(0, aBuf[0], bBuf[0]);
#pragma unroll
  for (int s = 0; s < S; ++s) {
    if (s + 1 < S) loadStep(s + 1, aBuf[(s + 1) & 1], bBuf[(s + 1) & 1]);
    domma(aBuf[s & 1], bBuf[s & 1]);
  }

  for (int nn = 0; nn < 2; ++nn) {
    const int col = nn ? col1 : col0;
    for (int t = 0; t < 4; ++t) {
      const int o0 = mBase + t * 16 + 8 * half;
      const long base = (long)col * C_ + o0;
      if constexpr (MODE == 0) {
        for (int r = 0; r < 8; ++r) outF[base + r] = acc[t][nn][r] + bias[o0 + r];
      } else if constexpr (MODE == 1) {
        v8us ob;
        for (int r = 0; r < 8; ++r) {
          float y = acc[t][nn][r] + bias[o0 + r];
          ob[r] = f2bf(y > 0.f ? y : 0.f);
        }
        *(v8us*)(outB + base) = ob;
      } else {
        for (int r = 0; r < 8; ++r) resid[base + r] += acc[t][nn][r] + bias[o0 + r];
      }
    }
  }
}

// ---------------------------------------------------------------------------
// Dilated conv as 5 shifted WMMA GEMM slabs, same tiling/pipeline.
// W: [KS][256][256] bf16 tap-major. out1 -> H f32 [N,L,C].
// ---------------------------------------------------------------------------
__global__ __launch_bounds__(256)
void gemm_conv(const u16* __restrict__ W, const u16* __restrict__ xt,
               const float* __restrict__ bias, float* __restrict__ outF, int d) {
  const int lane  = threadIdx.x & 31;
  const int wave  = threadIdx.x >> 5;
  const int half  = lane >> 4;
  const int mBase = (wave & 3) * 64;
  const int col0  = blockIdx.x * 64 + (wave >> 2) * 32 + (lane & 15);
  const int col1  = col0 + 16;
  const int l0    = col0 & (L_ - 1);
  const int l1    = col1 & (L_ - 1);
  const int mRow  = mBase + (lane & 15);

  v8f acc[4][2];
  for (int t = 0; t < 4; ++t)
    for (int nn = 0; nn < 2; ++nn)
      for (int r = 0; r < 8; ++r) acc[t][nn][r] = 0.f;

  auto loadStep = [&](int s, ABReg a[4], ABReg b[2]) {
    const int j  = s >> 3;          // 8 K-steps per tap
    const int kc = (s & 7) << 5;
    const int shift = (j - 2) * d;
    const bool v0 = ((unsigned)(l0 + shift)) < (unsigned)L_;
    const bool v1 = ((unsigned)(l1 + shift)) < (unsigned)L_;
    const u16* p0 = xt + (long)(col0 + shift) * C_ + kc;
    const u16* p1 = xt + (long)(col1 + shift) * C_ + kc;
    if (v0) {
      b[0].u[0] = *(const v8us*)(p0 + 16 * half);
      b[0].u[1] = *(const v8us*)(p0 + 16 * half + 8);
    } else {
      for (int i = 0; i < 8; ++i) { b[0].u[0][i] = 0; b[0].u[1][i] = 0; }
    }
    if (v1) {
      b[1].u[0] = *(const v8us*)(p1 + 16 * half);
      b[1].u[1] = *(const v8us*)(p1 + 16 * half + 8);
    } else {
      for (int i = 0; i < 8; ++i) { b[1].u[0][i] = 0; b[1].u[1][i] = 0; }
    }
    const u16* Aj = W + j * (C_ * C_);
    for (int t = 0; t < 4; ++t) {
      const u16* ar = Aj + (long)(mRow + t * 16) * C_ + kc;
      a[t].u[0] = *(const v8us*)(ar + 16 * half);
      a[t].u[1] = *(const v8us*)(ar + 16 * half + 8);
    }
  };
  auto domma = [&](ABReg a[4], ABReg b[2]) {
    for (int t = 0; t < 4; ++t) {
      acc[t][0] = wmma_bf16(a[t], b[0], acc[t][0]);
      acc[t][1] = wmma_bf16(a[t], b[1], acc[t][1]);
    }
  };

  constexpr int S = KS_ * (C_ >> 5);   // 40 steps
  ABReg aBuf[2][4], bBuf[2][2];
  loadStep(0, aBuf[0], bBuf[0]);
#pragma unroll
  for (int s = 0; s < S; ++s) {
    if (s + 1 < S) loadStep(s + 1, aBuf[(s + 1) & 1], bBuf[(s + 1) & 1]);
    domma(aBuf[s & 1], bBuf[s & 1]);
  }

  for (int nn = 0; nn < 2; ++nn) {
    const int col = nn ? col1 : col0;
    for (int t = 0; t < 4; ++t) {
      const int o0 = mBase + t * 16 + 8 * half;
      const long base = (long)col * C_ + o0;
      for (int r = 0; r < 8; ++r) outF[base + r] = acc[t][nn][r] + bias[o0 + r];
    }
  }
}

// ---------------------------------------------------------------------------
// Windowed attention: one wave per (n,l); wave32 __shfl_xor reductions.
// ---------------------------------------------------------------------------
__global__ __launch_bounds__(256)
void attn_kernel(const float* __restrict__ q, const float* __restrict__ k,
                 const float* __restrict__ v, float* __restrict__ h, int d) {
  const int wave = threadIdx.x >> 5;
  const int lane = threadIdx.x & 31;
  const int nl = blockIdx.x * 8 + wave;
  const int n = nl >> 12;
  const int l = nl & (L_ - 1);
  const long qbase = (long)nl * C_ + lane * 8;

  float qv[8];
  for (int i = 0; i < 8; ++i) qv[i] = q[qbase + i];

  float att[KS_];
  for (int j = 0; j < KS_; ++j) {
    const int lj = l + (j - 2) * d;
    const bool valid = ((unsigned)lj) < (unsigned)L_;
    float s = 0.f;
    if (valid) {
      const long kb = ((long)n * L_ + lj) * C_ + lane * 8;
      for (int i = 0; i < 8; ++i) s += qv[i] * k[kb + i];
    }
    for (int off = 16; off > 0; off >>= 1) s += __shfl_xor(s, off, 32);
    att[j] = valid ? (s * 0.0625f + 1e-6f) : -13.815511f;
  }
  float m = att[0];
  for (int j = 1; j < KS_; ++j) m = fmaxf(m, att[j]);
  float e[KS_], Z = 0.f;
  for (int j = 0; j < KS_; ++j) { e[j] = expf(att[j] - m); Z += e[j]; }
  const float inv = 1.f / Z;

  float racc[8] = {0.f, 0.f, 0.f, 0.f, 0.f, 0.f, 0.f, 0.f};
  for (int j = 0; j < KS_; ++j) {
    const int lj = l + (j - 2) * d;
    if (((unsigned)lj) < (unsigned)L_) {
      const float w = e[j] * inv;
      const long vb = ((long)n * L_ + lj) * C_ + lane * 8;
      for (int i = 0; i < 8; ++i) racc[i] += w * v[vb + i];
    }
  }
  for (int i = 0; i < 8; ++i) h[qbase + i] += racc[i];
}

// ---------------------------------------------------------------------------
// InstanceNorm1d over L per (n,c), in place; also emit bf16 copy.
// ---------------------------------------------------------------------------
__global__ __launch_bounds__(256)
void norm_kernel(float* __restrict__ h, u16* __restrict__ ht) {
  __shared__ float s1[256], s2[256];
  const int tid = threadIdx.x;
  const int n = blockIdx.x >> 8;
  const int c = blockIdx.x & 255;
  const long base = (long)n * L_ * C_ + c;
  float sum = 0.f, sq = 0.f;
  for (int l = tid; l < L_; l += 256) {
    float x = h[base + (long)l * C_];
    sum += x;
    sq += x * x;
  }
  s1[tid] = sum; s2[tid] = sq;
  __syncthreads();
  for (int st = 128; st > 0; st >>= 1) {
    if (tid < st) { s1[tid] += s1[tid + st]; s2[tid] += s2[tid + st]; }
    __syncthreads();
  }
  const float mu = s1[0] * (1.f / L_);
  const float var = s2[0] * (1.f / L_) - mu * mu;
  const float rs = rsqrtf(var + 1e-5f);
  for (int l = tid; l < L_; l += 256) {
    const long idx = base + (long)l * C_;
    float y = (h[idx] - mu) * rs;
    h[idx] = y;
    ht[idx] = f2bf(y);
  }
}

// ---------------------------------------------------------------------------
extern "C" void kernel_launch(void* const* d_in, const int* in_sizes, int n_in,
                              void* d_out, int out_size, void* d_ws, size_t ws_size,
                              hipStream_t stream) {
  (void)in_sizes; (void)n_in; (void)out_size; (void)ws_size;
  const float* x      = (const float*)d_in[0];
  const float* xcross = (const float*)d_in[1];
  const float* t_emb  = (const float*)d_in[2];
  const float* t_w    = (const float*)d_in[3];
  const float* t_b    = (const float*)d_in[4];
  const float* conv_w = (const float*)d_in[5];
  const float* conv_b = (const float*)d_in[6];
  const float* qw = (const float*)d_in[7];
  const float* qb = (const float*)d_in[8];
  const float* kw = (const float*)d_in[9];
  const float* kb = (const float*)d_in[10];
  const float* vw = (const float*)d_in[11];
  const float* vb = (const float*)d_in[12];
  const float* f1w = (const float*)d_in[13];
  const float* f1b = (const float*)d_in[14];
  const float* f2w = (const float*)d_in[15];
  const float* f2b = (const float*)d_in[16];

  // ---- workspace carve-up (bf16 region first, then f32 region) ----
  u16* Wc  = (u16*)d_ws;                       // 6*5*256*256
  u16* Wq  = Wc  + NLAY * KS_ * C_ * C_;       // 6*256*512
  u16* Wk  = Wq  + NLAY * C_ * (C_ + CC_);
  u16* Wv  = Wk  + NLAY * C_ * (C_ + CC_);     // 6*256*256
  u16* Wf1 = Wv  + NLAY * C_ * C_;
  u16* Wf2 = Wf1 + NLAY * C_ * C_;
  u16* xt  = Wf2 + NLAY * C_ * C_;             // NL*C bf16
  u16* xct = xt  + (long)NL_ * C_;
  u16* ht  = xct + (long)NL_ * CC_;
  u16* tt  = ht  + (long)NL_ * C_;
  float* X  = (float*)(tt + (long)NL_ * C_);   // NL*C f32 each
  float* Qb = X  + (long)NL_ * C_;
  float* Kf = Qb + (long)NL_ * C_;
  float* Vf = Kf + (long)NL_ * C_;
  float* H  = Vf + (long)NL_ * C_;

  const int EB = (NL_ * C_ + 255) / 256;

  time_kernel<<<N_ * C_, 256, 0, stream>>>(x, t_emb, t_w, t_b, X);
  xct_kernel<<<(N_ * CC_ * L_ + 255) / 256, 256, 0, stream>>>(xcross, xct);
  cvt_kernel<<<(NLAY * C_ * (C_ + CC_) + 255) / 256, 256, 0, stream>>>(qw, Wq, NLAY * C_ * (C_ + CC_));
  cvt_kernel<<<(NLAY * C_ * (C_ + CC_) + 255) / 256, 256, 0, stream>>>(kw, Wk, NLAY * C_ * (C_ + CC_));
  cvt_kernel<<<(NLAY * C_ * C_ + 255) / 256, 256, 0, stream>>>(vw, Wv, NLAY * C_ * C_);
  cvt_kernel<<<(NLAY * C_ * C_ + 255) / 256, 256, 0, stream>>>(f1w, Wf1, NLAY * C_ * C_);
  cvt_kernel<<<(NLAY * C_ * C_ + 255) / 256, 256, 0, stream>>>(f2w, Wf2, NLAY * C_ * C_);
  convw_kernel<<<(NLAY * KS_ * C_ * C_ + 255) / 256, 256, 0, stream>>>(conv_w, Wc);

  const int GB = NL_ / 64;   // 256 blocks, block tile 256(M) x 64(cols)

  for (int layer = 0; layer < NLAY; ++layer) {
    const int d = 1 << layer;
    cvt_kernel<<<EB, 256, 0, stream>>>(X, xt, NL_ * C_);
    gemm_conv<<<GB, 256, 0, stream>>>(Wc + (long)layer * KS_ * C_ * C_, xt,
                                      conv_b + layer * C_, H, d);
    gemm_pw<C_, CC_, 0><<<GB, 256, 0, stream>>>(
        Wq + (long)layer * C_ * (C_ + CC_), xt, xct, qb + layer * C_,
        Qb, nullptr, nullptr);
    gemm_pw<C_, CC_, 0><<<GB, 256, 0, stream>>>(
        Wk + (long)layer * C_ * (C_ + CC_), xt, xct, kb + layer * C_,
        Kf, nullptr, nullptr);
    gemm_pw<C_, 0, 0><<<GB, 256, 0, stream>>>(
        Wv + (long)layer * C_ * C_, xt, nullptr, vb + layer * C_,
        Vf, nullptr, nullptr);
    attn_kernel<<<NL_ / 8, 256, 0, stream>>>(Qb, Kf, Vf, H, d);
    norm_kernel<<<N_ * C_, 256, 0, stream>>>(H, ht);
    gemm_pw<C_, 0, 1><<<GB, 256, 0, stream>>>(
        Wf1 + (long)layer * C_ * C_, ht, nullptr, f1b + layer * C_,
        nullptr, tt, nullptr);
    gemm_pw<C_, 0, 2><<<GB, 256, 0, stream>>>(
        Wf2 + (long)layer * C_ * C_, tt, nullptr, f2b + layer * C_,
        nullptr, nullptr, X);
  }

  outT_kernel<<<(N_ * C_ * L_ + 255) / 256, 256, 0, stream>>>(X, (float*)d_out);
}
